// resnet18_69363721831046
// MI455X (gfx1250) — 8-bit matrix operand order repaired and checked by reading, compiled, not run on hardware
//
#include <hip/hip_runtime.h>
#include <cstdint>

typedef __attribute__((ext_vector_type(8))) int v8i;

#define EPSF 1e-5f


__global__ void pack_sign_kernel(const float* __restrict__ w, int8_t* __restrict__ wp,
                                 int Cin, int Cout, int K, int n) {
    int i = blockIdx.x * blockDim.x + threadIdx.x;
    if (i >= n) return;
    int co  = i % Cout;
    int r   = i / Cout;
    int ci  = r % Cin;
    int tap = r / Cin;
    wp[(size_t)co * K + tap * Cin + ci] = (w[i] >= 0.f) ? (int8_t)1 : (int8_t)-1;
}

__global__ void alpha_kernel(const float* __restrict__ w, int n, float* __restrict__ alpha) {
    __shared__ float sm[256];
    float s = 0.f;
    for (int i = threadIdx.x; i < n; i += 256) s += fabsf(w[i]);
    sm[threadIdx.x] = s;
    __syncthreads();
    for (int d = 128; d > 0; d >>= 1) {
        if (threadIdx.x < d) sm[threadIdx.x] += sm[threadIdx.x + d];
        __syncthreads();
    }
    if (threadIdx.x == 0) *alpha = sm[0] / (float)n;
}

__global__ void fold_bn_kernel(const float* __restrict__ m, const float* __restrict__ v,
                               const float* __restrict__ g, const float* __restrict__ b,
                               const float* __restrict__ alpha,
                               float* __restrict__ scale, float* __restrict__ bias, int c) {
    int i = blockIdx.x * blockDim.x + threadIdx.x;
    if (i >= c) return;
    float rs = rsqrtf(v[i] + EPSF);
    float a  = *alpha;
    scale[i] = g[i] * a * rs * (1.f / 3.f);
    bias[i]  = b[i] - g[i] * m[i] * rs;
}

__global__ void stem_kernel(const float* __restrict__ inp, const float* __restrict__ w,
                            const float* __restrict__ bm, const float* __restrict__ bv,
                            int8_t* __restrict__ out) {
    const int C = 64, P = 56, CH = 112;
    int i = blockIdx.x * blockDim.x + threadIdx.x;
    if (i >= 32 * P * P * C) return;
    int c = i % C;
    int t = i / C;
    int px = t % P; t /= P;
    int py = t % P;
    int b  = t / P;
    float mx = -1e30f;
    for (int wy = 0; wy < 3; ++wy) {
        int cy = 2 * py + wy;
        if (cy >= CH) continue;
        for (int wx = 0; wx < 3; ++wx) {
            int cx = 2 * px + wx;
            if (cx >= CH) continue;
            float s = 0.f;
            for (int ky = 0; ky < 7; ++ky) {
                int iy = 2 * cy + ky - 2;
                if ((unsigned)iy >= 224u) continue;
                for (int kx = 0; kx < 7; ++kx) {
                    int ix = 2 * cx + kx - 2;
                    if ((unsigned)ix >= 224u) continue;
                    const float* ip  = inp + (((size_t)b * 224 + iy) * 224 + ix) * 3;
                    const float* wp2 = w + ((ky * 7 + kx) * 3) * 64 + c;
                    s += ip[0] * wp2[0] + ip[1] * wp2[64] + ip[2] * wp2[128];
                }
            }
            mx = fmaxf(mx, s);
        }
    }
    float y = (mx - bm[c]) * rsqrtf(bv[c] + EPSF);
    y = fminf(fmaxf(y, 0.f), 1.f);
    out[i] = (int8_t)(int)(y * 3.f + 0.5f);
}

__global__ __launch_bounds__(128) void binconv_wmma_kernel(
        const int8_t* __restrict__ in, const int8_t* __restrict__ wp,
        const float* __restrict__ scale, const float* __restrict__ bias,
        int8_t* __restrict__ out,
        int Bsz, int H, int W, int Cin, int Cout,
        int OH, int OW, int stride, int ksz, int pad) {
    const int K    = ksz * ksz * Cin;
    const int lane = threadIdx.x & 31;
    const int wave = threadIdx.x >> 5;
    const int Mtot = Bsz * OH * OW;
    const int m_base = blockIdx.x * 64 + wave * 16;
    const int n0     = blockIdx.y * 32;
    const int row    = lane & 15;
    const int khalf  = lane >> 4;

    const int m = m_base + row;
    const bool mvalid = (m < Mtot);
    int b = 0, oy = 0, ox = 0;
    if (mvalid) {
        ox = m % OW;
        int t = m / OW;
        oy = t % OH;
        b  = t / OH;
    }

    const int* bptr0 = (const int*)(wp + (size_t)(n0 + row) * K) + khalf * 4;
    const int* bptr1 = (const int*)(wp + (size_t)(n0 + 16 + row) * K) + khalf * 4;

    v8i acc0 = {};
    v8i acc1 = {};
    const int oys = oy * stride - pad;
    const int oxs = ox * stride - pad;

    for (int ky = 0; ky < ksz; ++ky) {
        const int iy = oys + ky;
        for (int kx = 0; kx < ksz; ++kx) {
            const int ix = oxs + kx;
            const bool inb = mvalid && (unsigned)iy < (unsigned)H && (unsigned)ix < (unsigned)W;
            const int8_t* abase =
                in + (((size_t)b * H + iy) * W + ix) * (size_t)Cin + khalf * 16;
            for (int ci0 = 0; ci0 < Cin; ci0 += 64) {
                v8i a = {};
                if (inb) {
                    const int2* p = (const int2*)(abase + ci0);
                    int2 a0 = p[0], a1 = p[1], a2 = p[4], a3 = p[5];
                    a[0] = a0.x; a[1] = a0.y; a[2] = a1.x; a[3] = a1.y;
                    a[4] = a2.x; a[5] = a2.y; a[6] = a3.x; a[7] = a3.y;
                }
                __builtin_prefetch(bptr0 + 16, 0, 0);
                __builtin_prefetch(bptr1 + 16, 0, 0);
                const int4* q0 = (const int4*)bptr0;
                const int4* q1 = (const int4*)bptr1;
                int4 b0 = q0[0], b1 = q0[2];
                int4 c0 = q1[0], c1 = q1[2];
                v8i bm0, bm1;
                bm0[0] = b0.x; bm0[1] = b0.y; bm0[2] = b0.z; bm0[3] = b0.w;
                bm0[4] = b1.x; bm0[5] = b1.y; bm0[6] = b1.z; bm0[7] = b1.w;
                bm1[0] = c0.x; bm1[1] = c0.y; bm1[2] = c0.z; bm1[3] = c0.w;
                bm1[4] = c1.x; bm1[5] = c1.y; bm1[6] = c1.z; bm1[7] = c1.w;

                acc0 = __builtin_amdgcn_wmma_i32_16x16x64_iu8(false, a, true, bm0,
                                                              acc0, false, false);
                acc1 = __builtin_amdgcn_wmma_i32_16x16x64_iu8(false, a, true, bm1,
                                                              acc1, false, false);
                bptr0 += 16;
                bptr1 += 16;
            }
        }
    }

    const int col = lane & 15;
    const int c0i = n0 + col;
    const int c1i = n0 + 16 + col;
    const float s0 = scale[c0i], f0 = bias[c0i];
    const float s1 = scale[c1i], f1 = bias[c1i];
    for (int r = 0; r < 8; ++r) {
        int mrow = (lane < 16) ? r : (8 + r);
        int mm = m_base + mrow;
        if (mm < Mtot) {
            float y0 = fminf(fmaxf(s0 * (float)acc0[r] + f0, 0.f), 1.f);
            float y1 = fminf(fmaxf(s1 * (float)acc1[r] + f1, 0.f), 1.f);
            out[(size_t)mm * Cout + c0i] = (int8_t)(int)(y0 * 3.f + 0.5f);
            out[(size_t)mm * Cout + c1i] = (int8_t)(int)(y1 * 3.f + 0.5f);
        }
    }
}

__global__ void add_kernel(const int8_t* __restrict__ h, const int8_t* __restrict__ r,
                           int8_t* __restrict__ o, int n) {
    int i = blockIdx.x * blockDim.x + threadIdx.x;
    if (i < n) o[i] = (int8_t)(h[i] + r[i]);
}

__global__ void avgpool_kernel(const int8_t* __restrict__ x, float* __restrict__ pooled) {
    int i = blockIdx.x * blockDim.x + threadIdx.x;
    if (i >= 32 * 512) return;
    int c = i % 512, b = i / 512;
    int s = 0;
    for (int p = 0; p < 49; ++p) s += x[((size_t)b * 49 + p) * 512 + c];
    pooled[i] = (float)s / (3.f * 49.f);
}

__global__ void dense_kernel(const float* __restrict__ pooled, const float* __restrict__ w,
                             const float* __restrict__ scalu, float* __restrict__ out) {
    int i = blockIdx.x * blockDim.x + threadIdx.x;
    if (i >= 32 * 1000) return;
    int o = i % 1000, b = i / 1000;
    float s = 0.f;
    for (int c = 0; c < 512; ++c) {
        float p = pooled[b * 512 + c];
        s += (w[c * 1000 + o] >= 0.f) ? p : -p;
    }
    out[i] = s * (*scalu);
}

extern "C" void kernel_launch(void* const* d_in, const int* in_sizes, int n_in,
                              void* d_out, int out_size, void* d_ws, size_t ws_size,
                              hipStream_t stream) {
    (void)in_sizes; (void)n_in; (void)out_size; (void)ws_size;

    int cur = 0;
    const float* inputs  = (const float*)d_in[cur++];
    const float* conv1_w = (const float*)d_in[cur++];
    const float* bn1m = (const float*)d_in[cur++];
    const float* bn1v = (const float*)d_in[cur++];
    cur += 2;

    struct LayerP {
        const float *w, *m, *v, *g, *b;
        int cin, cout, ksz, stride;
    };
    LayerP L[19];
    int nl = 0;
    int li1[8], li2[8], lid[8];
    const int bf[8] = {64, 64, 128, 128, 256, 256, 512, 512};
    const int bs[8] = {1, 1, 2, 1, 2, 1, 2, 1};
    const int bd[8] = {0, 0, 1, 0, 1, 0, 1, 0};

    int cin = 64;
    for (int i = 0; i < 8; ++i) {
        int f = bf[i], s = bs[i];
        li1[i] = nl;
        L[nl++] = {(const float*)d_in[cur], (const float*)d_in[cur + 1],
                   (const float*)d_in[cur + 2], (const float*)d_in[cur + 3],
                   (const float*)d_in[cur + 4], cin, f, 3, s};
        cur += 5;
        li2[i] = nl;
        L[nl++] = {(const float*)d_in[cur], (const float*)d_in[cur + 1],
                   (const float*)d_in[cur + 2], (const float*)d_in[cur + 3],
                   (const float*)d_in[cur + 4], f, f, 3, 1};
        cur += 5;
        if (bd[i]) {
            lid[i] = nl;
            L[nl++] = {(const float*)d_in[cur], (const float*)d_in[cur + 1],
                       (const float*)d_in[cur + 2], (const float*)d_in[cur + 3],
                       (const float*)d_in[cur + 4], cin, f, 1, s};
            cur += 5;
        } else {
            lid[i] = -1;
        }
        cin = f;
    }
    const float* dense_w = (const float*)d_in[cur++];
    const float* scalu   = (const float*)d_in[cur++];

    char* wsb  = (char*)d_ws;
    size_t off = 0;
    auto carve = [&](size_t bytes) -> void* {
        off = (off + 255) & ~(size_t)255;
        void* p = wsb + off;
        off += bytes;
        return p;
    };
    const size_t ACT = (size_t)32 * 56 * 56 * 64;
    int8_t* P[5];
    for (int i = 0; i < 5; ++i) P[i] = (int8_t*)carve(ACT);

    int8_t* WP[19]; float* SC[19]; float* BI[19]; float* AL[19];
    for (int l = 0; l < nl; ++l) {
        int K = L[l].ksz * L[l].ksz * L[l].cin;
        WP[l] = (int8_t*)carve((size_t)L[l].cout * K);
        SC[l] = (float*)carve((size_t)L[l].cout * 4);
        BI[l] = (float*)carve((size_t)L[l].cout * 4);
        AL[l] = (float*)carve(4);
    }
    float* pooled = (float*)carve((size_t)32 * 512 * 4);

    for (int l = 0; l < nl; ++l) {
        int K = L[l].ksz * L[l].ksz * L[l].cin;
        int n = K * L[l].cout;
        pack_sign_kernel<<<(n + 255) / 256, 256, 0, stream>>>(L[l].w, WP[l], L[l].cin,
                                                              L[l].cout, K, n);
        alpha_kernel<<<1, 256, 0, stream>>>(L[l].w, n, AL[l]);
        fold_bn_kernel<<<(L[l].cout + 255) / 256, 256, 0, stream>>>(
            L[l].m, L[l].v, L[l].g, L[l].b, AL[l], SC[l], BI[l], L[l].cout);
    }

    {
        int n0 = 32 * 56 * 56 * 64;
        stem_kernel<<<(n0 + 255) / 256, 256, 0, stream>>>(inputs, conv1_w, bn1m, bn1v, P[0]);
        hipMemcpyAsync(P[1], P[0], (size_t)n0, hipMemcpyDeviceToDevice, stream);
    }

    int8_t* x   = P[0];
    int8_t* res = P[1];
    int8_t* pool[3] = {P[2], P[3], P[4]};
    int np = 3;
    auto bpop  = [&]() { return pool[--np]; };
    auto bpush = [&](int8_t* p) { pool[np++] = p; };

    auto conv = [&](int l, const int8_t* src, int8_t* dst, int H, int W, int OH, int OW) {
        int Mtot = 32 * OH * OW;
        dim3 grid((Mtot + 63) / 64, L[l].cout / 32);
        binconv_wmma_kernel<<<grid, 128, 0, stream>>>(
            src, WP[l], SC[l], BI[l], dst, 32, H, W, L[l].cin, L[l].cout, OH, OW,
            L[l].stride, L[l].ksz, (L[l].ksz == 3) ? 1 : 0);
    };

    int h = 56, w = 56;
    for (int i = 0; i < 8; ++i) {
        int f = bf[i], s = bs[i];
        int oh = (s == 2) ? h / 2 : h;
        int ow = (s == 2) ? w / 2 : w;

        int8_t* h1 = bpop();
        conv(li1[i], x, h1, h, w, oh, ow);
        int8_t* h2 = bpop();
        conv(li2[i], h1, h2, oh, ow, oh, ow);
        bpush(h1);

        if (bd[i]) {
            int8_t* rd = bpop();
            conv(lid[i], res, rd, h, w, oh, ow);
            bpush(res);
            res = rd;
        }

        int8_t* xn = bpop();
        int n = 32 * oh * ow * f;
        add_kernel<<<(n + 255) / 256, 256, 0, stream>>>(h2, res, xn, n);
        bpush(x);
        bpush(res);
        x   = xn;
        res = h2;
        h = oh;
        w = ow;
    }

    avgpool_kernel<<<(32 * 512 + 255) / 256, 256, 0, stream>>>(x, pooled);
    dense_kernel<<<(32 * 1000 + 255) / 256, 256, 0, stream>>>(pooled, dense_w, scalu,
                                                              (float*)d_out);
}
